// DHYPR_86002425135140
// MI455X (gfx1250) — compile-verified, matrix operand order checked by reading, not run on hardware
//
#include <hip/hip_runtime.h>
#include <hip/hip_bf16.h>

typedef __attribute__((ext_vector_type(16))) __bf16 v16bf;
typedef __attribute__((ext_vector_type(8)))  __bf16 v8bf;
typedef __attribute__((ext_vector_type(8)))  float  v8f;

#define MIN_NORM_F 1e-15f
#define PROJ_EPS_F 1e-5f

__device__ __forceinline__ float wsum(float v) {
#pragma unroll
  for (int o = 16; o > 0; o >>= 1) v += __shfl_xor(v, o, 32);
  return v;
}

__device__ __forceinline__ float artanh_clip(float x) {
  x = fminf(fmaxf(x, -1.0f + 1e-7f), 1.0f - 1e-7f);
  return 0.5f * (log1pf(x) - log1pf(-x));
}

// ---------------------------------------------------------------------------
// x -> proj(expmap0(x)) as bf16, plus analytic row norm. One block per row.
// ---------------------------------------------------------------------------
__global__ void __launch_bounds__(256)
prep_x_kernel(const float* __restrict__ x, __bf16* __restrict__ xh,
              float* __restrict__ xnorm, int Nn) {
  const int row = blockIdx.x;
  const int t = threadIdx.x;
  float v = x[(size_t)row * 256 + t];
  __shared__ float part[8];
  __shared__ float stot;
  float ps = wsum(v * v);
  if ((t & 31) == 0) part[t >> 5] = ps;
  __syncthreads();
  if (t == 0) {
    float s = 0.f;
#pragma unroll
    for (int i = 0; i < 8; ++i) s += part[i];
    stot = s;
  }
  __syncthreads();
  const float maxn = 1.0f - PROJ_EPS_F;
  float n = fmaxf(sqrtf(stot), MIN_NORM_F);
  float th = tanhf(n);
  float s = th / n;
  if (th > maxn) s = maxn / n;
  xh[(size_t)row * 256 + t] = (__bf16)(v * s);
  if (t == 0) xnorm[row] = fminf(th, maxn);
}

// ---------------------------------------------------------------------------
// hb[b][layer] = proj(expmap0(bias)). 8 waves, one bias vector each.
// ---------------------------------------------------------------------------
__global__ void __launch_bounds__(256)
prep_bias(const float* __restrict__ b1, const float* __restrict__ b2,
          float* __restrict__ hb) {
  const int wave = threadIdx.x >> 5;   // 0..7 : branch*2 + layer
  const int lane = threadIdx.x & 31;
  const int branch = wave >> 1, layer = wave & 1;
  const float* srcp = (layer == 0 ? b1 : b2) + branch * 64 + lane * 2;
  float2 v = { srcp[0], srcp[1] };
  float n = fmaxf(sqrtf(wsum(v.x * v.x + v.y * v.y)), MIN_NORM_F);
  const float maxn = 1.0f - PROJ_EPS_F;
  float th = tanhf(n);
  float s = th / n;
  if (th > maxn) s = maxn / n;
  float* d = hb + wave * 64 + lane * 2;
  d[0] = s * v.x;
  d[1] = s * v.y;
}

// ---------------------------------------------------------------------------
// HypLinear (mobius_matvec -> proj -> mobius_add(bias) -> proj) fused with
// logmap0.  WMMA bf16 GEMM: block computes a 32x64 tile (8 waves of 16x16).
//   hin   : [Nn, F] bf16 (row-major input point on the ball)
//   xnorm : [Nn]    analytic norms of hin rows
//   W     : [64, F] f32 (converted to bf16 in LDS; B columns == W rows)
//   hb    : [64]    hyperbolic bias vector
//   ht    : [Nn,64] f32 output tangent features
// ---------------------------------------------------------------------------
__global__ void __launch_bounds__(256)
hyp_linear_wmma(const __bf16* __restrict__ hin, const float* __restrict__ xnorm,
                const float* __restrict__ W, const float* __restrict__ hb,
                float* __restrict__ ht, int Nn, int F) {
  __shared__ __align__(16) __bf16 Wl[64 * 256];
  __shared__ float mvt[32 * 64];
  const int tid = threadIdx.x;
  const int nw = 64 * F;
  for (int i = tid; i < nw; i += 256) Wl[i] = (__bf16)W[i];
  __syncthreads();

  const int wave = tid >> 5, lane = tid & 31;
  const int lrow = lane & 15;
  const int klo = (lane >> 4) * 8;                 // K sub-offset per ISA layout
  const int rowBase = blockIdx.x * 32 + (wave >> 2) * 16;
  const int colBase = (wave & 3) * 16;
  int arow = rowBase + lrow;
  if (arow >= Nn) arow = Nn - 1;                   // keep EXEC all-ones for WMMA
  const __bf16* aptr = hin + (size_t)arow * F;
  const __bf16* bptr = Wl + (size_t)(colBase + lrow) * F;

  v8f acc = {};
  for (int kb = 0; kb < F; kb += 32) {
    v8bf alo = *reinterpret_cast<const v8bf*>(aptr + kb + klo);
    v8bf ahi = *reinterpret_cast<const v8bf*>(aptr + kb + 16 + klo);
    v8bf blo = *reinterpret_cast<const v8bf*>(bptr + kb + klo);
    v8bf bhi = *reinterpret_cast<const v8bf*>(bptr + kb + 16 + klo);
    v16bf av = __builtin_shufflevector(alo, ahi, 0,1,2,3,4,5,6,7,8,9,10,11,12,13,14,15);
    v16bf bv = __builtin_shufflevector(blo, bhi, 0,1,2,3,4,5,6,7,8,9,10,11,12,13,14,15);
    acc = __builtin_amdgcn_wmma_f32_16x16x32_bf16(false, av, false, bv,
                                                  (short)0, acc, false, false);
  }
  // C/D layout: vgpr r, lane L -> M = r + 8*(L>>4), N = L&15
  const int mrow = (wave >> 2) * 16 + (lane >> 4) * 8;
  const int mcol = colBase + lrow;
#pragma unroll
  for (int r = 0; r < 8; ++r) mvt[(mrow + r) * 64 + mcol] = acc[r];
  __syncthreads();

  // Fused epilogue: one wave per row, 4 rows per wave over the 32-row tile.
  const float maxn = 1.0f - PROJ_EPS_F;
  float2 hbv = { hb[lane * 2], hb[lane * 2 + 1] };
  float y2 = wsum(hbv.x * hbv.x + hbv.y * hbv.y);
  for (int it = 0; it < 4; ++it) {
    const int lr = it * 8 + wave;
    const int grow = blockIdx.x * 32 + lr;
    float2 m = { mvt[lr * 64 + lane * 2], mvt[lr * 64 + lane * 2 + 1] };
    float mxn = sqrtf(wsum(m.x * m.x + m.y * m.y));
    const int gi = grow < Nn ? grow : Nn - 1;
    float xn = fmaxf(xnorm[gi], MIN_NORM_F);
    // mobius_matvec (c=1): res = tanh(mxn/xn * artanh(xn)) * mx / mxn
    float arg = (fmaxf(mxn, MIN_NORM_F) / xn) * artanh_clip(xn);
    float th = tanhf(arg);
    float s = th / fmaxf(mxn, MIN_NORM_F);
    if (th > maxn) s = maxn / fmaxf(mxn, MIN_NORM_F);   // fused proj
    float xnr = fminf(th, maxn);
    if (mxn < 1e-20f) { s = 0.f; xnr = 0.f; }           // all-zero mx branch
    float2 xv = { s * m.x, s * m.y };
    float x2 = xnr * xnr;
    // mobius_add(xv, hb)
    float xy = wsum(xv.x * hbv.x + xv.y * hbv.y);
    float k1 = 1.f + 2.f * xy + y2;
    float k2 = 1.f - x2;
    float den = fmaxf(1.f + 2.f * xy + x2 * y2, MIN_NORM_F);
    float2 r2 = { (k1 * xv.x + k2 * hbv.x) / den, (k1 * xv.y + k2 * hbv.y) / den };
    // proj + logmap0
    float rn = sqrtf(wsum(r2.x * r2.x + r2.y * r2.y));
    float pf = (rn > maxn) ? (maxn / rn) : 1.f;
    r2.x *= pf; r2.y *= pf;
    float rc = fminf(rn, maxn);
    float ls = artanh_clip(rc) / fmaxf(rc, MIN_NORM_F);
    if (grow < Nn) {
      ht[(size_t)grow * 64 + lane * 2]     = ls * r2.x;
      ht[(size_t)grow * 64 + lane * 2 + 1] = ls * r2.y;
    }
  }
}

// ---------------------------------------------------------------------------
__global__ void __launch_bounds__(256)
zero_f32(float* __restrict__ p, int n) {
  int i = blockIdx.x * 256 + threadIdx.x;
  if (i < n) p[i] = 0.f;
}

// ---------------------------------------------------------------------------
// agg[dst] += v * ht[src] : one wave per edge, 2 floats per lane.
// ---------------------------------------------------------------------------
__global__ void __launch_bounds__(256)
spmm_edges(const float* __restrict__ ht, const float* __restrict__ vals,
           const int* __restrict__ src, const int* __restrict__ dst,
           float* __restrict__ agg, int E) {
  const int e = blockIdx.x * 8 + (threadIdx.x >> 5);
  if (e >= E) return;
  const int lane = threadIdx.x & 31;
  const int s = src[e], d = dst[e];
  const float v = vals[e];
  const float* hs = ht + (size_t)s * 64 + lane * 2;
  float2 f = { hs[0], hs[1] };
  float* ad = agg + (size_t)d * 64 + lane * 2;
  atomicAdd(ad,     v * f.x);
  atomicAdd(ad + 1, v * f.y);
}

// ---------------------------------------------------------------------------
// proj(expmap0(agg)) -> relu(logmap0) -> proj(expmap0) : one wave per row.
// Optionally also writes bf16 copy + analytic norm for the next layer.
// ---------------------------------------------------------------------------
__global__ void __launch_bounds__(256)
post_agg(const float* __restrict__ agg, float* __restrict__ outf,
         __bf16* __restrict__ outb, float* __restrict__ onorm, int Nn) {
  const int row = (int)((blockIdx.x * 256 + threadIdx.x) >> 5);
  if (row >= Nn) return;
  const int lane = threadIdx.x & 31;
  const float maxn = 1.0f - PROJ_EPS_F;
  const float* a = agg + (size_t)row * 64 + lane * 2;
  float2 u = { a[0], a[1] };
  float un = fmaxf(sqrtf(wsum(u.x * u.x + u.y * u.y)), MIN_NORM_F);
  float th = tanhf(un);
  float s = th / un;
  if (th > maxn) s = maxn / un;
  float2 h = { s * u.x, s * u.y };
  float hn = fminf(th, maxn);
  float l = artanh_clip(hn) / fmaxf(hn, MIN_NORM_F);
  float2 t = { fmaxf(l * h.x, 0.f), fmaxf(l * h.y, 0.f) };  // relu in tangent
  float tn = fmaxf(sqrtf(wsum(t.x * t.x + t.y * t.y)), MIN_NORM_F);
  float th2 = tanhf(tn);
  float s2 = th2 / tn;
  if (th2 > maxn) s2 = maxn / tn;
  float2 o = { s2 * t.x, s2 * t.y };
  float* of = outf + (size_t)row * 64 + lane * 2;
  of[0] = o.x; of[1] = o.y;
  if (outb) {
    __bf16* ob = outb + (size_t)row * 64 + lane * 2;
    ob[0] = (__bf16)o.x; ob[1] = (__bf16)o.y;
    if (lane == 0 && onorm) onorm[row] = fminf(th2, maxn);
  }
}

// ---------------------------------------------------------------------------
// mobius_mulscaler(1/8) per branch, mobius_add chain, mean of 5 logmaps,
// expmap0+proj.  One wave per row.
// ---------------------------------------------------------------------------
__global__ void __launch_bounds__(256)
combine_out(const float* __restrict__ finals, float* __restrict__ out, int Nn) {
  const int row = (int)((blockIdx.x * 256 + threadIdx.x) >> 5);
  if (row >= Nn) return;
  const int lane = threadIdx.x & 31;
  const float maxn = 1.0f - PROJ_EPS_F;
  float2 f[4]; float nb[4];
#pragma unroll
  for (int b = 0; b < 4; ++b) {
    const float* p = finals + ((size_t)b * Nn + row) * 64 + lane * 2;
    f[b].x = p[0]; f[b].y = p[1];
    nb[b] = sqrtf(wsum(f[b].x * f[b].x + f[b].y * f[b].y));
  }
  float2 t = { 0.f, 0.f };
#pragma unroll
  for (int b = 0; b < 4; ++b) {
    float nc = fmaxf(nb[b], MIN_NORM_F);
    float wn = tanhf(0.125f * artanh_clip(nc));
    float sw = wn / nc;
    float2 w = { sw * f[b].x, sw * f[b].y };
    if (b == 0) {
      t = w;
    } else {
      float x2 = wsum(t.x * t.x + t.y * t.y);
      float y2 = wn * wn;
      float xy = wsum(t.x * w.x + t.y * w.y);
      float k1 = 1.f + 2.f * xy + y2;
      float k2 = 1.f - x2;
      float den = fmaxf(1.f + 2.f * xy + x2 * y2, MIN_NORM_F);
      t.x = (k1 * t.x + k2 * w.x) / den;
      t.y = (k1 * t.y + k2 * w.y) / den;
    }
  }
  float tn = sqrtf(wsum(t.x * t.x + t.y * t.y));
  float2 acc = { 0.f, 0.f };
#pragma unroll
  for (int b = 0; b < 4; ++b) {
    float nc = fmaxf(nb[b], MIN_NORM_F);
    float ls = artanh_clip(nc) / nc;
    acc.x += ls * f[b].x; acc.y += ls * f[b].y;
  }
  {
    float nc = fmaxf(tn, MIN_NORM_F);
    float ls = artanh_clip(nc) / nc;
    acc.x += ls * t.x; acc.y += ls * t.y;
  }
  acc.x *= 0.2f; acc.y *= 0.2f;
  float an = fmaxf(sqrtf(wsum(acc.x * acc.x + acc.y * acc.y)), MIN_NORM_F);
  float th = tanhf(an);
  float s = th / an;
  if (th > maxn) s = maxn / an;
  out[(size_t)row * 64 + lane * 2]     = s * acc.x;
  out[(size_t)row * 64 + lane * 2 + 1] = s * acc.y;
}

// ---------------------------------------------------------------------------
extern "C" void kernel_launch(void* const* d_in, const int* in_sizes, int n_in,
                              void* d_out, int out_size, void* d_ws, size_t ws_size,
                              hipStream_t stream) {
  (void)n_in; (void)out_size; (void)ws_size;
  const float* x    = (const float*)d_in[0];
  const float* W1   = (const float*)d_in[1];
  const float* b1   = (const float*)d_in[2];
  const float* W2   = (const float*)d_in[3];
  const float* b2   = (const float*)d_in[4];
  const float* vals = (const float*)d_in[5];
  const int*   src  = (const int*)d_in[6];
  const int*   dst  = (const int*)d_in[7];
  const int Nn = in_sizes[0] / 256;
  const int E  = in_sizes[5] / 4;

  char* ws = (char*)d_ws;
  size_t off = 0;
  auto carve = [&](size_t bytes) -> char* {
    char* p = ws + off;
    off += (bytes + 255) & ~(size_t)255;
    return p;
  };
  __bf16* xhyp   = (__bf16*)carve((size_t)Nn * 256 * sizeof(__bf16));
  float*  xn0    = (float*) carve((size_t)Nn * sizeof(float));
  float*  xn1    = (float*) carve((size_t)Nn * sizeof(float));
  float*  hb     = (float*) carve(8 * 64 * sizeof(float));
  float*  finals = (float*) carve((size_t)4 * Nn * 64 * sizeof(float));
  float*  ht     = (float*) carve((size_t)Nn * 64 * sizeof(float));
  float*  agg    = (float*) carve((size_t)Nn * 64 * sizeof(float));
  __bf16* h1b    = (__bf16*)carve((size_t)Nn * 64 * sizeof(__bf16));

  prep_x_kernel<<<Nn, 256, 0, stream>>>(x, xhyp, xn0, Nn);
  prep_bias<<<1, 256, 0, stream>>>(b1, b2, hb);

  const int gemmGrid = (Nn + 31) / 32;
  const int rowGrid  = (Nn + 7) / 8;
  const int zeroGrid = (Nn * 64 + 255) / 256;
  const int spmmGrid = (E + 7) / 8;

  for (int b = 0; b < 4; ++b) {
    // ---- layer 1 (F = 256) ----
    hyp_linear_wmma<<<gemmGrid, 256, 0, stream>>>(
        xhyp, xn0, W1 + (size_t)b * 64 * 256, hb + (b * 2 + 0) * 64, ht, Nn, 256);
    zero_f32<<<zeroGrid, 256, 0, stream>>>(agg, Nn * 64);
    spmm_edges<<<spmmGrid, 256, 0, stream>>>(
        ht, vals + (size_t)b * E, src + (size_t)b * E, dst + (size_t)b * E, agg, E);
    post_agg<<<rowGrid, 256, 0, stream>>>(agg, ht, h1b, xn1, Nn);
    // ---- layer 2 (F = 64) ----
    hyp_linear_wmma<<<gemmGrid, 256, 0, stream>>>(
        h1b, xn1, W2 + (size_t)b * 64 * 64, hb + (b * 2 + 1) * 64, ht, Nn, 64);
    zero_f32<<<zeroGrid, 256, 0, stream>>>(agg, Nn * 64);
    spmm_edges<<<spmmGrid, 256, 0, stream>>>(
        ht, vals + (size_t)b * E, src + (size_t)b * E, dst + (size_t)b * E, agg, E);
    post_agg<<<rowGrid, 256, 0, stream>>>(
        agg, finals + (size_t)b * Nn * 64, nullptr, nullptr, Nn);
  }
  combine_out<<<rowGrid, 256, 0, stream>>>(finals, (float*)d_out, Nn);
}
